// Mamba2SSD_76854144795254
// MI455X (gfx1250) — matrix operand order repaired and checked by reading, not compiled, not run on hardware
//
#include <hip/hip_runtime.h>
#include <hip/hip_bf16.h>
#include <math.h>
#include <stdint.h>

typedef _Float16 v16h __attribute__((ext_vector_type(16)));
typedef _Float16 h8_t __attribute__((ext_vector_type(8)));
typedef _Float16 h4_t __attribute__((ext_vector_type(4)));
typedef float    v8f  __attribute__((ext_vector_type(8)));

#define D_MODEL   2048
#define CONV_DIM  6144
#define N_HEADS   32
#define SEQ       4096
#define BATCH     2
#define M_TOTAL   (BATCH*SEQ)   /* 8192 rows */

// Build a 16-half WMMA fragment from two 16-byte LDS chunks (two ds_load_b128).
__device__ __forceinline__ v16h ld_frag16(const _Float16* p0, const _Float16* p1) {
  h8_t lo = *(const h8_t*)p0;
  h8_t hi = *(const h8_t*)p1;
  v16h r;
#pragma unroll
  for (int i = 0; i < 8; ++i) { r[i] = lo[i]; r[i + 8] = hi[i]; }
  return r;
}

// CDNA5 async copy: global -> LDS, 16 bytes per lane, tracked by ASYNCcnt.
// LDS byte address = low 32 bits of the generic pointer (flat aperture layout).
__device__ __forceinline__ void async_copy_b128(uint32_t lds_addr, const _Float16* gptr) {
  asm volatile("global_load_async_to_lds_b128 %0, %1, off"
               :: "v"(lds_addr), "v"((uint64_t)(uintptr_t)gptr)
               : "memory");
}

// ---------------------------------------------------------------------------
// fp32 -> f16 conversion pass (vectorized, grid-stride)
// ---------------------------------------------------------------------------
__global__ __launch_bounds__(256) void cvt_f16_kernel(
    const float* __restrict__ S, _Float16* __restrict__ D, int n4)
{
  int i = blockIdx.x * 256 + threadIdx.x;
  const int stride = gridDim.x * 256;
  for (; i < n4; i += stride) {
    float4 v = ((const float4*)S)[i];
    h4_t h = {(_Float16)v.x, (_Float16)v.y, (_Float16)v.z, (_Float16)v.w};
    ((h4_t*)D)[i] = h;
  }
}

// ---------------------------------------------------------------------------
// f16 GEMM: O = A16 @ B16^T.  A16: Mx2048 row-major, B16: Nx2048 row-major.
// 128x128 block tile, BK=32, 8 waves (2x4), each wave 64x32 via 4x2 WMMA.
// Double-buffered LDS filled with global_load_async_to_lds_b128.
// MODE 0: split epilogue (cols<2048 -> O0 = z, else O1 = xBC_raw)
// MODE 1: plain epilogue to O0.
// ---------------------------------------------------------------------------
template<int MODE>
__global__ __launch_bounds__(256) void gemm_f16_kernel(
    const _Float16* __restrict__ A16, const _Float16* __restrict__ B16,
    float* __restrict__ O0, float* __restrict__ O1)
{
  __shared__ __align__(16) _Float16 As[2][128 * 32];
  __shared__ __align__(16) _Float16 Bs[2][128 * 32];
  const int bm = blockIdx.y, bn = blockIdx.x;
  const int tid = threadIdx.x;
  const int lane = tid & 31, wave = tid >> 5;
  const int hf = lane >> 4, lm = lane & 15;
  const int wm = wave >> 2, wn = wave & 3;          // 2x4 wave grid
  v8f acc[4][2] = {};
  const uint32_t a_base = (uint32_t)(uintptr_t)&As[0][0];
  const uint32_t b_base = (uint32_t)(uintptr_t)&Bs[0][0];

  // Stage 0 into buffer 0 (each wave issues exactly 4 async instructions).
#pragma unroll
  for (int p = 0; p < 2; ++p) {
    int c = tid + p * 256;                 // 512 x 16B chunks per 8KB tile
    int rr = c >> 2, cc = (c & 3) * 8;     // row, half-offset within 32-wide row
    async_copy_b128(a_base + (uint32_t)(rr * 64 + cc * 2),
                    A16 + (size_t)(bm * 128 + rr) * 2048 + cc);
    async_copy_b128(b_base + (uint32_t)(rr * 64 + cc * 2),
                    B16 + (size_t)(bn * 128 + rr) * 2048 + cc);
  }

  for (int ks = 0; ks < 64; ++ks) {
    const int buf = ks & 1;
    if (ks + 1 < 64) {
      const int k0 = (ks + 1) * 32;
      const int boff = (buf ^ 1) * 8192;   // byte offset of other buffer
#pragma unroll
      for (int p = 0; p < 2; ++p) {
        int c = tid + p * 256;
        int rr = c >> 2, cc = (c & 3) * 8;
        async_copy_b128(a_base + (uint32_t)(boff + rr * 64 + cc * 2),
                        A16 + (size_t)(bm * 128 + rr) * 2048 + k0 + cc);
        async_copy_b128(b_base + (uint32_t)(boff + rr * 64 + cc * 2),
                        B16 + (size_t)(bn * 128 + rr) * 2048 + k0 + cc);
      }
      asm volatile("s_wait_asynccnt 4" ::: "memory");   // stage ks landed
    } else {
      asm volatile("s_wait_asynccnt 0" ::: "memory");
    }
    __syncthreads();                        // publish stage ks block-wide

    const _Float16* Ab = &As[buf][0];
    const _Float16* Bb = &Bs[buf][0];
    v16h af[4], bf[2];
#pragma unroll
    for (int mt = 0; mt < 4; ++mt) {
      int row = wm * 64 + mt * 16 + lm;
      af[mt] = ld_frag16(&Ab[row * 32 + hf * 8], &Ab[row * 32 + 16 + hf * 8]);
    }
#pragma unroll
    for (int nt = 0; nt < 2; ++nt) {
      int col = wn * 32 + nt * 16 + lm;
      bf[nt] = ld_frag16(&Bb[col * 32 + hf * 8], &Bb[col * 32 + 16 + hf * 8]);
    }
#pragma unroll
    for (int mt = 0; mt < 4; ++mt)
#pragma unroll
      for (int nt = 0; nt < 2; ++nt)
        acc[mt][nt] = __builtin_amdgcn_wmma_f32_16x16x32_f16(
            false, af[mt], false, bf[nt], (short)0, acc[mt][nt], false, false);
    __syncthreads();                        // safe to overwrite buf next iter
  }

  // ---- epilogue: block-uniform branching only -----------------------------
  if (MODE == 0 && bn >= 16) {              // xBC columns
#pragma unroll
    for (int mt = 0; mt < 4; ++mt)
#pragma unroll
      for (int nt = 0; nt < 2; ++nt) {
        int colg = bn * 128 + wn * 32 + nt * 16 + lm - 2048;
#pragma unroll
        for (int r = 0; r < 8; ++r) {
          int rowg = bm * 128 + wm * 64 + mt * 16 + hf * 8 + r;
          O1[(size_t)rowg * 6144 + colg] = acc[mt][nt][r];
        }
      }
  } else {                                  // z columns (MODE 0) or plain out
#pragma unroll
    for (int mt = 0; mt < 4; ++mt)
#pragma unroll
      for (int nt = 0; nt < 2; ++nt) {
        int colg = bn * 128 + wn * 32 + nt * 16 + lm;
#pragma unroll
        for (int r = 0; r < 8; ++r) {
          int rowg = bm * 128 + wm * 64 + mt * 16 + hf * 8 + r;
          O0[(size_t)rowg * 2048 + colg] = acc[mt][nt][r];
        }
      }
  }
}

// ---------------------------------------------------------------------------
// Depthwise conv (k=3, pad=1) along L + bias + SiLU
// ---------------------------------------------------------------------------
__global__ __launch_bounds__(256) void conv_silu_kernel(
    const float* __restrict__ X, const float* __restrict__ Wc,
    const float* __restrict__ Bc, float* __restrict__ O)
{
  const int ch = blockIdx.x * 256 + threadIdx.x;
  const int b  = blockIdx.z;
  const int l0 = blockIdx.y * 512;
  const float w0 = Wc[ch * 3 + 0], w1 = Wc[ch * 3 + 1], w2 = Wc[ch * 3 + 2];
  const float bias = Bc[ch];
  const float* in  = X + (size_t)b * SEQ * CONV_DIM + ch;
  float*       out = O + (size_t)b * SEQ * CONV_DIM + ch;
  float xm = (l0 > 0) ? in[(size_t)(l0 - 1) * CONV_DIM] : 0.f;
  float xc = in[(size_t)l0 * CONV_DIM];
  for (int l = l0; l < l0 + 512; ++l) {
    float xp = (l + 1 < SEQ) ? in[(size_t)(l + 1) * CONV_DIM] : 0.f;
    float v = w0 * xm + w1 * xc + w2 * xp + bias;
    out[(size_t)l * CONV_DIM] = v / (1.f + expf(-v));   // SiLU
    xm = xc; xc = xp;
  }
}

// ---------------------------------------------------------------------------
// chunk_states[b,h,c,n] = sum_l x[l,n]*B[l,n]
// ---------------------------------------------------------------------------
__global__ __launch_bounds__(64) void chunk_state_kernel(
    const float* __restrict__ XBCa, float* __restrict__ CS)
{
  const int b = blockIdx.z, h = blockIdx.y, c = blockIdx.x;
  const int n = threadIdx.x;
  const size_t base = ((size_t)(b * SEQ + c * 64)) * CONV_DIM + h * 64 + n;
  float acc = 0.f;
  for (int l = 0; l < 64; ++l) {
    float xv = XBCa[base + (size_t)l * CONV_DIM];
    float bv = XBCa[base + (size_t)l * CONV_DIM + 2048];
    acc += xv * bv;
  }
  CS[(((size_t)(b * N_HEADS + h)) * 64 + c) * 64 + n] = acc;
}

// ---------------------------------------------------------------------------
// Inter-chunk scan: prev[c] = states[c-1]; states = states*exp(63*A_log) + cs
// ---------------------------------------------------------------------------
__global__ __launch_bounds__(64) void scan_kernel(
    const float* __restrict__ Alog, const float* __restrict__ CS,
    float* __restrict__ PS)
{
  const int bh = blockIdx.x;
  const int h  = bh & 31;
  const int n  = threadIdx.x;
  const float d = expf(63.f * Alog[h]);
  float prev = 0.f;
  const size_t base = (size_t)bh * 64 * 64 + n;
  for (int c = 0; c < 64; ++c) {
    PS[base + (size_t)c * 64] = prev;
    prev = prev * d + CS[base + (size_t)c * 64];
  }
}

// ---------------------------------------------------------------------------
// SSD per (b,h,c): G = (C·B^T)*causal ; Y = G·x + (prev·C)[l] ; gate by SiLU(z)
// ---------------------------------------------------------------------------
__global__ __launch_bounds__(128) void ssd_y_kernel(
    const float* __restrict__ XBCa, const float* __restrict__ PrevS,
    const float* __restrict__ Z, float* __restrict__ Y)
{
  __shared__ __align__(16) _Float16 Cs [64 * 64];  // C row-major (A operand)
  __shared__ __align__(16) _Float16 Bsm[64 * 64];  // B row-major ([n=m][k=n])
  __shared__ __align__(16) _Float16 Xst[64 * 64];  // x transposed ([n=p][k=m])
  __shared__ __align__(16) _Float16 Gs [64 * 64];  // masked G (A operand)
  __shared__ float Cf[64 * 64];
  __shared__ float prevS[64];
  __shared__ float yoff[64];
  const int b = blockIdx.z, h = blockIdx.y, c = blockIdx.x;
  const int tid = threadIdx.x;
  const int lane = tid & 31, w = tid >> 5;
  const int hf = lane >> 4, lm = lane & 15;
  const size_t base = ((size_t)(b * SEQ + c * 64)) * CONV_DIM + h * 64;
#pragma unroll
  for (int i = 0; i < 8; ++i) {
    int j = tid + i * 128;                  // float4 index 0..1023
    int l = j >> 4, p4 = (j & 15) * 4;
    float4 xv = *(const float4*)(XBCa + base + (size_t)l * CONV_DIM + p4);
    float4 bv = *(const float4*)(XBCa + base + (size_t)l * CONV_DIM + 2048 + p4);
    float4 cv = *(const float4*)(XBCa + base + (size_t)l * CONV_DIM + 4096 + p4);
    h4_t hb = {(_Float16)bv.x, (_Float16)bv.y, (_Float16)bv.z, (_Float16)bv.w};
    *(h4_t*)&Bsm[l * 64 + p4] = hb;
    h4_t hc = {(_Float16)cv.x, (_Float16)cv.y, (_Float16)cv.z, (_Float16)cv.w};
    *(h4_t*)&Cs[l * 64 + p4] = hc;
    *(float4*)&Cf[l * 64 + p4] = cv;
    Xst[(p4 + 0) * 64 + l] = (_Float16)xv.x;
    Xst[(p4 + 1) * 64 + l] = (_Float16)xv.y;
    Xst[(p4 + 2) * 64 + l] = (_Float16)xv.z;
    Xst[(p4 + 3) * 64 + l] = (_Float16)xv.w;
  }
  if (tid < 64) prevS[tid] = PrevS[(((size_t)(b * N_HEADS + h)) * 64 + c) * 64 + tid];
  __syncthreads();

  const int rowl = w * 16 + lm;
  v8f g[4] = {};
#pragma unroll
  for (int s = 0; s < 2; ++s) {
    int k0 = s * 32;
    v16h a = ld_frag16(&Cs[rowl * 64 + k0 + hf * 8], &Cs[rowl * 64 + k0 + 16 + hf * 8]);
#pragma unroll
    for (int nt = 0; nt < 4; ++nt) {
      int m = nt * 16 + lm;
      v16h bb = ld_frag16(&Bsm[m * 64 + k0 + hf * 8], &Bsm[m * 64 + k0 + 16 + hf * 8]);
      g[nt] = __builtin_amdgcn_wmma_f32_16x16x32_f16(false, a, false, bb, (short)0,
                                                     g[nt], false, false);
    }
  }
  // causal mask in f32 accumulators, re-stage as f16 A operand
#pragma unroll
  for (int nt = 0; nt < 4; ++nt)
#pragma unroll
    for (int r = 0; r < 8; ++r) {
      int l = w * 16 + hf * 8 + r;
      int m = nt * 16 + lm;
      float v = (m <= l) ? g[nt][r] : 0.f;
      Gs[l * 64 + m] = (_Float16)v;
    }
  if (tid < 64) {               // Y_off[l] = prev . C[l,:]
    float acc = 0.f;
#pragma unroll 8
    for (int n = 0; n < 64; ++n) acc += prevS[n] * Cf[tid * 64 + n];
    yoff[tid] = acc;
  }
  __syncthreads();

  v8f y[4] = {};
#pragma unroll
  for (int s = 0; s < 2; ++s) {
    int k0 = s * 32;
    v16h a = ld_frag16(&Gs[rowl * 64 + k0 + hf * 8], &Gs[rowl * 64 + k0 + 16 + hf * 8]);
#pragma unroll
    for (int nt = 0; nt < 4; ++nt) {
      int p = nt * 16 + lm;
      v16h bb = ld_frag16(&Xst[p * 64 + k0 + hf * 8], &Xst[p * 64 + k0 + 16 + hf * 8]);
      y[nt] = __builtin_amdgcn_wmma_f32_16x16x32_f16(false, a, false, bb, (short)0,
                                                     y[nt], false, false);
    }
  }
#pragma unroll
  for (int nt = 0; nt < 4; ++nt)
#pragma unroll
    for (int r = 0; r < 8; ++r) {
      int l = w * 16 + hf * 8 + r;
      int p = nt * 16 + lm;
      size_t idx = ((size_t)(b * SEQ + c * 64 + l)) * 2048 + h * 64 + p;
      float zv = Z[idx];
      float sz = zv / (1.f + expf(-zv));
      Y[idx] = (y[nt][r] + yoff[l]) * sz;
    }
}

// ---------------------------------------------------------------------------
// RMSNorm per row + cast to f16 for the output GEMM
// ---------------------------------------------------------------------------
__global__ __launch_bounds__(256) void rms_kernel(
    const float* __restrict__ Ys, const float* __restrict__ Wrms,
    _Float16* __restrict__ Yn)
{
  __shared__ float red[256];
  const size_t row = blockIdx.x;
  const int tid = threadIdx.x;
  float s = 0.f;
#pragma unroll
  for (int i = 0; i < 8; ++i) {
    float v = Ys[row * 2048 + tid + i * 256];
    s += v * v;
  }
  red[tid] = s; __syncthreads();
  for (int off = 128; off > 0; off >>= 1) {
    if (tid < off) red[tid] += red[tid + off];
    __syncthreads();
  }
  float scale = rsqrtf(red[0] * (1.f / 2048.f) + 1.1920929e-07f);
#pragma unroll
  for (int i = 0; i < 8; ++i) {
    int d = tid + i * 256;
    float v = Ys[row * 2048 + d];
    Yn[row * 2048 + d] = (_Float16)(v * scale * Wrms[d]);
  }
}

// ---------------------------------------------------------------------------
extern "C" void kernel_launch(void* const* d_in, const int* in_sizes, int n_in,
                              void* d_out, int out_size, void* d_ws, size_t ws_size,
                              hipStream_t stream) {
  (void)in_sizes; (void)n_in; (void)out_size; (void)ws_size;
  const float* u      = (const float*)d_in[0];
  const float* W_in   = (const float*)d_in[1];
  const float* W_out  = (const float*)d_in[2];
  const float* conv_w = (const float*)d_in[3];
  const float* conv_b = (const float*)d_in[4];
  const float* A_log  = (const float*)d_in[5];
  const float* rms_w  = (const float*)d_in[6];

  char* ws = (char*)d_ws;
  size_t off = 0;
  float* xbc_raw  = (float*)(ws + off); off += (size_t)M_TOTAL * CONV_DIM * 4;
  float* zbuf     = (float*)(ws + off); off += (size_t)M_TOTAL * D_MODEL * 4;
  float* xbc_act  = (float*)(ws + off); off += (size_t)M_TOTAL * CONV_DIM * 4;
  float* cs       = (float*)(ws + off); off += (size_t)BATCH * N_HEADS * 64 * 64 * 4;
  float* ps       = (float*)(ws + off); off += (size_t)BATCH * N_HEADS * 64 * 64 * 4;
  float* ysc      = (float*)(ws + off); off += (size_t)M_TOTAL * D_MODEL * 4;
  _Float16* yn    = (_Float16*)(ws + off); off += (size_t)M_TOTAL * D_MODEL * 2;
  _Float16* u16   = (_Float16*)(ws + off); off += (size_t)M_TOTAL * D_MODEL * 2;
  _Float16* win16 = (_Float16*)(ws + off); off += (size_t)8192 * D_MODEL * 2;
  _Float16* wout16= (_Float16*)(ws + off); off += (size_t)D_MODEL * D_MODEL * 2;

  // One-time fp32 -> f16 conversion of GEMM operands (bandwidth-trivial).
  cvt_f16_kernel<<<2048, 256, 0, stream>>>(u,     u16,    M_TOTAL * D_MODEL / 4);
  cvt_f16_kernel<<<2048, 256, 0, stream>>>(W_in,  win16,  8192 * D_MODEL / 4);
  cvt_f16_kernel<<<1024, 256, 0, stream>>>(W_out, wout16, D_MODEL * D_MODEL / 4);

  // zxbcdt = u @ W_in^T (dt columns skipped): split into z / xBC_raw
  gemm_f16_kernel<0><<<dim3(64, 64), 256, 0, stream>>>(u16, win16, zbuf, xbc_raw);
  conv_silu_kernel<<<dim3(24, 8, 2), 256, 0, stream>>>(xbc_raw, conv_w, conv_b, xbc_act);
  chunk_state_kernel<<<dim3(64, 32, 2), 64, 0, stream>>>(xbc_act, cs);
  scan_kernel<<<64, 64, 0, stream>>>(A_log, cs, ps);
  ssd_y_kernel<<<dim3(64, 32, 2), 128, 0, stream>>>(xbc_act, ps, zbuf, ysc);
  rms_kernel<<<8192, 256, 0, stream>>>(ysc, rms_w, yn);
  // out = y_norm @ W_out^T
  gemm_f16_kernel<1><<<dim3(16, 64), 256, 0, stream>>>(yn, wout16, (float*)d_out, nullptr);
}
